// LSTMModel_39513699123284
// MI455X (gfx1250) — compile-verified, matrix operand order checked by reading, not run on hardware
//
#include <hip/hip_runtime.h>
#include <hip/hip_bf16.h>

// ---------------------------------------------------------------------------
// LSTM on MI455X (gfx1250): bf16 WMMA GEMMs with double-buffered
// GLOBAL_LOAD_ASYNC_TO_LDS staging, fused LSTM cell + O=1 head.
// 1.65 TFLOP recurrent math; 12 MB bf16 weights L2-resident across 256 steps.
// ---------------------------------------------------------------------------

#define BB 512   // batch
#define SS 256   // sequence
#define II 16    // input dim
#define EE 512   // embedding dim
#define HH 1024  // hidden dim

typedef __attribute__((ext_vector_type(16))) __bf16 v16bf;
typedef __attribute__((ext_vector_type(8)))  float  v8f;

// workspace layout (bytes)
#define WIH_OFF  0                         // [4H][E] bf16 : 4 MB
#define WHH_OFF  (4*1024*1024)             // [4H][H] bf16 : 8 MB
#define H_OFF    (12*1024*1024)            // [2][B][H] bf16 ping-pong : 2 MB
#define C_OFF    (14*1024*1024)            // [B][H] f32 : 2 MB
#define EMB_OFF  (16*1024*1024)            // [S][B][E] bf16 : 128 MB

#if defined(__has_builtin)
#if __has_builtin(__builtin_amdgcn_global_load_async_to_lds_b128) && \
    __has_builtin(__builtin_amdgcn_s_wait_asynccnt)
#define USE_ASYNC 1
#endif
#endif

// The async builtin's pointer params are GCC-style int4 vectors in
// global (as1) / LDS (as3) address spaces.
typedef int gvec4i __attribute__((vector_size(16)));
typedef __attribute__((address_space(1))) gvec4i* gas4p;
typedef __attribute__((address_space(3))) gvec4i* las4p;

__device__ __forceinline__ unsigned short f2bf(float f) {
    unsigned int u = __float_as_uint(f);
    u += 0x7FFFu + ((u >> 16) & 1u);          // round-to-nearest-even
    return (unsigned short)(u >> 16);
}

__device__ __forceinline__ float sigmf(float x) {
    return 1.0f / (1.0f + __expf(-x));
}

// 16B global -> LDS copy. Async (ASYNCcnt-tracked) when the toolchain exposes
// the gfx1250 builtin; otherwise a register relay (still LDS-staged).
__device__ __forceinline__ void cp16(const unsigned short* g, unsigned short* l) {
#ifdef USE_ASYNC
    __builtin_amdgcn_global_load_async_to_lds_b128(
        (gas4p)(unsigned long long)g,
        (las4p)(unsigned int)(unsigned long long)l,  // low 32b of flat = LDS offset
        0, 0);
#else
    *reinterpret_cast<uint4*>(l) = *reinterpret_cast<const uint4*>(g);
#endif
}

__device__ __forceinline__ void stage_sync() {
#ifdef USE_ASYNC
    __builtin_amdgcn_s_wait_asynccnt(0);
#endif
    __syncthreads();
}

// LDS rows padded to 40 bf16 (80 B) -> the 16 rows of a fragment hit 16
// distinct bank groups (stride 20 words, gcd(20,64)=4, period 16).
#define LDSPITCH 40

// Build a 16x32 bf16 WMMA fragment from a padded row-major LDS tile,
// matching the CDNA5 16-bit A/B^T VGPR layout:
//   lanes 0-15 : v0..3 = K[k0..k0+7],  v4..7 = K[k0+16..k0+23]   (row = lane)
//   lanes16-31 : v0..3 = K[k0+8..+15], v4..7 = K[k0+24..+31]
__device__ __forceinline__ v16bf lds_frag(const unsigned short* buf, int row, int lhi8) {
    const unsigned short* p = buf + row * LDSPITCH + lhi8;
    union { uint4 u[2]; v16bf v; } c;
    c.u[0] = *reinterpret_cast<const uint4*>(p);
    c.u[1] = *reinterpret_cast<const uint4*>(p + 16);
    return c.v;
}

#define WMMA_BF16(a, b, c) __builtin_amdgcn_wmma_f32_16x16x32_bf16( \
    false, (a), false, (b), (short)0, (c), false, false)

// ---------------------------------------------------------------------------
// Kernel 1: weights -> bf16, zero h/c, out = b_out
// ---------------------------------------------------------------------------
#define N_WIH (4*HH*EE)
#define N_WHH (4*HH*HH)
#define N_H   (2*BB*HH)
#define N_C   (BB*HH)
#define N_OUT (BB*SS)
#define N_PREP (N_WIH + N_WHH + N_H + N_C + N_OUT)

__global__ void prep_kernel(const float* __restrict__ W_ih, const float* __restrict__ W_hh,
                            const float* __restrict__ b_out,
                            unsigned short* __restrict__ wih_b, unsigned short* __restrict__ whh_b,
                            unsigned short* __restrict__ hbuf, float* __restrict__ cbuf,
                            float* __restrict__ out) {
    size_t idx = (size_t)blockIdx.x * blockDim.x + threadIdx.x;
    if (idx < N_WIH) { wih_b[idx] = f2bf(W_ih[idx]); return; }
    idx -= N_WIH;
    if (idx < N_WHH) { whh_b[idx] = f2bf(W_hh[idx]); return; }
    idx -= N_WHH;
    if (idx < N_H)   { hbuf[idx] = 0; return; }
    idx -= N_H;
    if (idx < N_C)   { cbuf[idx] = 0.0f; return; }
    idx -= N_C;
    if (idx < N_OUT) { out[idx] = b_out[0]; }
}

// ---------------------------------------------------------------------------
// Kernel 2: emb[s][b][e] = bf16(relu(x[b,s,:] . W_emb[e,:] + b_emb[e]))
// I=16 -> trivial VALU GEMM (2.1 GFLOP); stored time-major so every step's
// A-matrix rows are K-contiguous.
// ---------------------------------------------------------------------------
__global__ void emb_kernel(const float* __restrict__ x, const float* __restrict__ W_emb,
                           const float* __restrict__ b_emb, unsigned short* __restrict__ emb) {
    size_t idx = (size_t)blockIdx.x * blockDim.x + threadIdx.x;   // [0, S*B*E)
    int e  = (int)(idx & (EE - 1));
    int sb = (int)(idx >> 9);            // s*B + b
    int s  = sb >> 9;                    // B = 512
    int b  = sb & (BB - 1);
    const float* xr = x + ((size_t)b * SS + s) * II;
    const float* wr = W_emb + (size_t)e * II;
    float acc = b_emb[e];
#pragma unroll
    for (int i = 0; i < II; ++i) acc = fmaf(xr[i], wr[i], acc);
    emb[idx] = f2bf(fmaxf(acc, 0.0f));
}

// ---------------------------------------------------------------------------
// Kernel 3: one LSTM step.
//   grid (B/64, H/64), 256 threads = 8 waves.
//   Per K=32 chunk the WG async-copies A(64x32) and B(256x32 = 64 cols x 4
//   gates) into LDS (double-buffered, 2 x 25.6 KB), then each wave computes
//   8 accumulators: 2 M-sub-tiles x 4 gates over its 16 columns
//   (8 wmma per chunk, 48 chunks: K = E(512) then H(1024)).
//   Fused: c' = sig(f)*c + sig(i)*tanh(g); h' = sig(o)*tanh(c');
//   out[b,s] += relu(h') . W_out  via 16-lane shuffle reduce + f32 atomic.
// ---------------------------------------------------------------------------
#define NCHUNK 48
#define ATILE  (64 * LDSPITCH)            // 2560 ushort
#define BTILE  (256 * LDSPITCH)           // 10240 ushort
#define BUFSZ  (ATILE + BTILE)            // 12800 ushort = 25.6 KB

__global__ __launch_bounds__(256)
void lstm_step_kernel(const unsigned short* __restrict__ emb,   // [S][B][E]
                      const unsigned short* __restrict__ wih,   // [4H][E]
                      const unsigned short* __restrict__ whh,   // [4H][H]
                      unsigned short* __restrict__ hbuf,        // [2][B][H]
                      float* __restrict__ cbuf,                 // [B][H]
                      const float* __restrict__ b_ih, const float* __restrict__ b_hh,
                      const float* __restrict__ w_out,          // [H]
                      float* __restrict__ out,                  // [B][S]
                      int s) {
    __shared__ unsigned short smem[2][BUFSZ];                   // 51.2 KB

    const int tid  = threadIdx.x;
    const int lane = tid & 31;
    const int wave = tid >> 5;
    const int m_half = wave & 1;        // 32-row half of the 64-row tile
    const int n_grp  = wave >> 1;       // 16-col group of the 64-col tile

    const int m0 = blockIdx.x * 64;     // batch rows
    const int j0 = blockIdx.y * 64;     // hidden cols

    const int lrow = lane & 15;
    const int lhi8 = (lane >> 4) << 3;  // 0 or 8 (K-half select)

    const unsigned short* emb_s  = emb + (size_t)s * BB * EE;
    const unsigned short* h_prev = hbuf + (size_t)(s & 1) * BB * HH;
    unsigned short*       h_next = hbuf + (size_t)((s + 1) & 1) * BB * HH;

    // stage one K=32 chunk: 1280 x b128 transfers, 5 per thread,
    // statically split A/B so there is no EXEC divergence.
    auto stage = [&](int chunk, unsigned short* buf) {
        const unsigned short* asrc; const unsigned short* wsrc; int ld, k;
        if (chunk < EE / 32) { asrc = emb_s;  wsrc = wih; ld = EE; k = chunk << 5; }
        else                 { asrc = h_prev; wsrc = whh; ld = HH; k = (chunk - EE / 32) << 5; }
        const int c8 = (tid & 3) << 3;
        {   // A tile: 64 rows x 4 x b128 (1 per thread)
            int row = tid >> 2;
            cp16(asrc + (size_t)(m0 + row) * ld + k + c8,
                 buf + row * LDSPITCH + c8);
        }
#pragma unroll
        for (int i = 0; i < 4; ++i) {   // B tile: 256 rows x 4 x b128 (4 per thread)
            int r = (tid + (i << 8)) >> 2;        // 0..255
            int q = r >> 6, c = r & 63;           // gate, col-in-tile
            cp16(wsrc + (size_t)(q * HH + j0 + c) * ld + k + c8,
                 buf + ATILE + r * LDSPITCH + c8);
        }
    };

    v8f acc[2][4] = {};                 // [m-sub][gate]

    // preload all 6 fragments (one ds clause), then an 8-wmma burst
    auto compute = [&](const unsigned short* buf) {
        const unsigned short* bufA = buf;
        const unsigned short* bufB = buf + ATILE;
        v16bf a0 = lds_frag(bufA, m_half * 32 + lrow,      lhi8);
        v16bf a1 = lds_frag(bufA, m_half * 32 + 16 + lrow, lhi8);
        v16bf b0 = lds_frag(bufB, 0 * 64 + n_grp * 16 + lrow, lhi8);
        v16bf b1 = lds_frag(bufB, 1 * 64 + n_grp * 16 + lrow, lhi8);
        v16bf b2 = lds_frag(bufB, 2 * 64 + n_grp * 16 + lrow, lhi8);
        v16bf b3 = lds_frag(bufB, 3 * 64 + n_grp * 16 + lrow, lhi8);
        acc[0][0] = WMMA_BF16(a0, b0, acc[0][0]);
        acc[1][0] = WMMA_BF16(a1, b0, acc[1][0]);
        acc[0][1] = WMMA_BF16(a0, b1, acc[0][1]);
        acc[1][1] = WMMA_BF16(a1, b1, acc[1][1]);
        acc[0][2] = WMMA_BF16(a0, b2, acc[0][2]);
        acc[1][2] = WMMA_BF16(a1, b2, acc[1][2]);
        acc[0][3] = WMMA_BF16(a0, b3, acc[0][3]);
        acc[1][3] = WMMA_BF16(a1, b3, acc[1][3]);
    };

    stage(0, smem[0]);
    stage_sync();
#pragma unroll 1
    for (int ch = 0; ch < NCHUNK; ch += 2) {   // ping-pong, compile-time buffers
        stage(ch + 1, smem[1]);
        compute(smem[0]);
        stage_sync();
        if (ch + 2 < NCHUNK) stage(ch + 2, smem[0]);
        compute(smem[1]);
        stage_sync();
    }

    // fused LSTM cell + output head
    const int jg = j0 + n_grp * 16 + lrow;      // this lane's column
    float bi[4];
#pragma unroll
    for (int q = 0; q < 4; ++q) bi[q] = b_ih[q * HH + jg] + b_hh[q * HH + jg];
    const float wo = w_out[jg];
    const int m_off = (lane >> 4) << 3;         // C-layout: rows r / r+8

#pragma unroll
    for (int ms = 0; ms < 2; ++ms) {
        const int mrow0 = m0 + m_half * 32 + ms * 16;
#pragma unroll
        for (int r = 0; r < 8; ++r) {
            const int mg = mrow0 + r + m_off;
            const size_t cell = (size_t)mg * HH + jg;
            float ig = sigmf(acc[ms][0][r] + bi[0]);
            float fg = sigmf(acc[ms][1][r] + bi[1]);
            float gg = tanhf(acc[ms][2][r] + bi[2]);
            float og = sigmf(acc[ms][3][r] + bi[3]);
            float c2 = fg * cbuf[cell] + ig * gg;
            float h2 = og * tanhf(c2);
            cbuf[cell] = c2;
            h_next[cell] = f2bf(h2);
            float p = fmaxf(h2, 0.0f) * wo;     // O=1 head partial
            p += __shfl_xor(p, 1);
            p += __shfl_xor(p, 2);
            p += __shfl_xor(p, 4);
            p += __shfl_xor(p, 8);
            if ((lane & 15) == 0)
                atomicAdd(&out[(size_t)mg * SS + s], p);
        }
    }
}

// ---------------------------------------------------------------------------
extern "C" void kernel_launch(void* const* d_in, const int* in_sizes, int n_in,
                              void* d_out, int out_size, void* d_ws, size_t ws_size,
                              hipStream_t stream) {
    const float* x     = (const float*)d_in[0];
    const float* W_emb = (const float*)d_in[1];
    const float* b_emb = (const float*)d_in[2];
    const float* W_ih  = (const float*)d_in[3];
    const float* b_ih  = (const float*)d_in[4];
    const float* W_hh  = (const float*)d_in[5];
    const float* b_hh  = (const float*)d_in[6];
    const float* W_out = (const float*)d_in[7];
    const float* b_out = (const float*)d_in[8];
    float* out = (float*)d_out;

    char* ws = (char*)d_ws;
    unsigned short* wih_b = (unsigned short*)(ws + WIH_OFF);
    unsigned short* whh_b = (unsigned short*)(ws + WHH_OFF);
    unsigned short* hbuf  = (unsigned short*)(ws + H_OFF);
    float*          cbuf  = (float*)(ws + C_OFF);
    unsigned short* embb  = (unsigned short*)(ws + EMB_OFF);

    {   // 1) weight conversion + state/output init
        int blocks = (N_PREP + 255) / 256;
        prep_kernel<<<blocks, 256, 0, stream>>>(W_ih, W_hh, b_out,
                                                wih_b, whh_b, hbuf, cbuf, out);
    }
    {   // 2) embedding (time-major bf16)
        size_t total = (size_t)SS * BB * EE;
        int blocks = (int)(total / 256);
        emb_kernel<<<blocks, 256, 0, stream>>>(x, W_emb, b_emb, embb);
    }
    // 3) 256 sequential LSTM steps
    dim3 grid(BB / 64, HH / 64);
    for (int s = 0; s < SS; ++s) {
        lstm_step_kernel<<<grid, 256, 0, stream>>>(embb, wih_b, whh_b, hbuf, cbuf,
                                                   b_ih, b_hh, W_out, out, s);
    }
}
